// CachedTransformerBlock_57561151701242
// MI455X (gfx1250) — compile-verified, matrix operand order checked by reading, not run on hardware
//
#include <hip/hip_runtime.h>
#include <hip/hip_bf16.h>

// ---------- types / helpers ----------
typedef __attribute__((ext_vector_type(16))) __bf16 v16bf;
typedef __attribute__((ext_vector_type(8)))  float  v8f;

union FragBF { unsigned u[8]; v16bf v; };

__device__ __forceinline__ unsigned bf16_rn(float f) {
    unsigned u = __float_as_uint(f);
    return (u + 0x7FFFu + ((u >> 16) & 1u)) >> 16;   // round-to-nearest-even
}
__device__ __forceinline__ unsigned pack_bf16(float lo, float hi) {
    return bf16_rn(lo) | (bf16_rn(hi) << 16);
}
__device__ __forceinline__ unsigned short bf16_h(float f) {
    return (unsigned short)bf16_rn(f);
}

#define WMMA_BF16(A, B, C) \
    __builtin_amdgcn_wmma_f32_16x16x32_bf16(false, (A), false, (B), (short)0, (C), false, false)

// Problem constants
#define DMODEL 2048
#define NHEADS 16
#define HDIM   128
#define DFF    8192
#define BB     8
#define TT     16
#define SCACHE 4096
#define STOT   4112            // SCACHE + TT
#define ROWS   128             // B*T tokens

// ---------- 1) cache -> output concat copy (pure bandwidth) ----------
__global__ __launch_bounds__(256) void copy_cache_kernel(
    const float4* __restrict__ ks, const float4* __restrict__ vs,
    float4* __restrict__ kd, float4* __restrict__ vd)
{
    int i = blockIdx.x * 256 + threadIdx.x;        // 0 .. 16777215 (per tensor)
    int row = i >> 5;                              // (b*H+h)*4096 + s
    int c   = i & 31;                              // float4 within 128-d row
    int bh  = row >> 12;
    int s   = row & 4095;
    int o   = (bh * STOT + s) * 32 + c;
    kd[o] = ks[i];
    vd[o] = vs[i];
}

// ---------- 2) LayerNorm (fp32 in -> bf16 out) ----------
__global__ __launch_bounds__(256) void ln_kernel(
    const float* __restrict__ x, const float* __restrict__ g,
    const float* __restrict__ b, unsigned short* __restrict__ out)
{
    __shared__ float red[256];
    __shared__ float s_mean, s_rstd;
    int row = blockIdx.x, tid = threadIdx.x;
    const float* xr = x + (long)row * DMODEL;

    float s = 0.f;
    for (int i = tid; i < DMODEL; i += 256) s += xr[i];
    red[tid] = s; __syncthreads();
    for (int off = 128; off >= 1; off >>= 1) {
        if (tid < off) red[tid] += red[tid + off];
        __syncthreads();
    }
    if (tid == 0) s_mean = red[0] * (1.f / DMODEL);
    __syncthreads();
    float mean = s_mean;

    float vv = 0.f;
    for (int i = tid; i < DMODEL; i += 256) { float d = xr[i] - mean; vv += d * d; }
    red[tid] = vv; __syncthreads();
    for (int off = 128; off >= 1; off >>= 1) {
        if (tid < off) red[tid] += red[tid + off];
        __syncthreads();
    }
    if (tid == 0) s_rstd = rsqrtf(red[0] * (1.f / DMODEL) + 1e-5f);
    __syncthreads();
    float rstd = s_rstd;

    for (int i = tid; i < DMODEL; i += 256)
        out[(long)row * DMODEL + i] = bf16_h((xr[i] - mean) * rstd * g[i] + b[i]);
}

// ---------- 3) generic M=128 GEMM via v_wmma_f32_16x16x32_bf16 ----------
// C[128,N] = A_bf16[128,K] x W_f32[K,N]  (+bias)(+GELU)(+resid) -> f32 and/or bf16
// K-chunk = 64. Both operands staged in LDS:
//   As[row][k]   bf16 (A chunk, 16 KB)
//   Wt[col][k]   bf16 (W tile transposed, converted ONCE per block, 2 KB)
#define GEMM_KC 64

__global__ __launch_bounds__(256) void gemm128_kernel(
    const unsigned short* __restrict__ A, const float* __restrict__ W,
    const float* __restrict__ bias, const float* __restrict__ resid,
    float* __restrict__ outF, unsigned short* __restrict__ outB,
    int K, int N, int act)
{
    __shared__ unsigned short As[ROWS * GEMM_KC];  // 16 KB
    __shared__ unsigned short Wt[16 * GEMM_KC];    // 2 KB   [col][k]
    const int tid  = threadIdx.x;
    const int lane = tid & 31;
    const int wave = tid >> 5;
    const int l16  = lane & 15;
    const int khalf = (lane >= 16) ? 8 : 0;
    const int n0  = blockIdx.x * 16;
    const int col = n0 + l16;
    const int arow = wave * 16 + l16;
    const int wkr  = tid >> 2;     // 0..63 : k row for W staging
    const int wcs  = tid & 3;      // 0..3  : 4-col segment for W staging

    v8f acc = {0.f, 0.f, 0.f, 0.f, 0.f, 0.f, 0.f, 0.f};

    for (int k0 = 0; k0 < K; k0 += GEMM_KC) {
        // coalesced prefetch of the next W tile toward L2 (global_prefetch_b8)
        if (k0 + GEMM_KC < K)
            __builtin_prefetch(W + (long)(k0 + GEMM_KC + wkr) * N + n0 + wcs * 4, 0, 1);

        // stage 128x64 bf16 A chunk: 1024 x uint4, 4 per thread
        #pragma unroll
        for (int it = 0; it < 4; ++it) {
            int u = it * 256 + tid;
            int row = u >> 3, seg = u & 7;
            *(uint4*)(&As[row * GEMM_KC + seg * 8]) =
                *(const uint4*)(A + (long)row * K + k0 + seg * 8);
        }
        // stage 64x16 W tile: one coalesced float4 per thread, convert once,
        // store transposed so k-pairs are contiguous for B fragments
        {
            float4 wf = *(const float4*)(W + (long)(k0 + wkr) * N + n0 + wcs * 4);
            Wt[(wcs * 4 + 0) * GEMM_KC + wkr] = bf16_h(wf.x);
            Wt[(wcs * 4 + 1) * GEMM_KC + wkr] = bf16_h(wf.y);
            Wt[(wcs * 4 + 2) * GEMM_KC + wkr] = bf16_h(wf.z);
            Wt[(wcs * 4 + 3) * GEMM_KC + wkr] = bf16_h(wf.w);
        }
        __syncthreads();

        #pragma unroll
        for (int kh = 0; kh < 2; ++kh) {
            FragBF fa, fb;
            #pragma unroll
            for (int v = 0; v < 8; ++v) {
                int kk = ((v >= 4) ? 16 : 0) + khalf + (v & 3) * 2;
                fa.u[v] = *(const unsigned*)(&As[arow * GEMM_KC + kh * 32 + kk]);
                fb.u[v] = *(const unsigned*)(&Wt[l16 * GEMM_KC + kh * 32 + kk]);
            }
            acc = WMMA_BF16(fa.v, fb.v, acc);
        }
        __syncthreads();
    }

    #pragma unroll
    for (int v = 0; v < 8; ++v) {
        int row = wave * 16 + v + ((lane >= 16) ? 8 : 0);
        float val = acc[v];
        if (bias)  val += bias[col];
        if (act == 1) val = 0.5f * val * (1.0f + erff(val * 0.70710678118654752f));
        if (resid) val += resid[(long)row * N + col];
        if (outF)  outF[(long)row * N + col] = val;
        if (outB)  outB[(long)row * N + col] = bf16_h(val);
    }
}

// ---------- 4) scatter K_new / V_new into output position s = 4096+t ----------
__global__ __launch_bounds__(256) void scatter_new_kernel(
    const float* __restrict__ knew, const float* __restrict__ vnew,
    float* __restrict__ Kout, float* __restrict__ Vout)
{
    int i = blockIdx.x * 256 + threadIdx.x;        // 0 .. 262143
    int row = i >> 11, col = i & 2047;             // [token][h*128+d]
    int b = row >> 4, t = row & 15;
    int h = col >> 7, d = col & 127;
    long o = ((long)((b * NHEADS + h) * STOT + SCACHE + t)) * HDIM + d;
    Kout[o] = knew[i];
    Vout[o] = vnew[i];
}

// ---------- 5) flash attention, one block per (b,h), WMMA scores + WMMA P.V ----------
#define ATT_CK 64   // key-chunk

__global__ __launch_bounds__(256) void attn_kernel(
    const float* __restrict__ q_ws, const float* __restrict__ Kout,
    const float* __restrict__ Vout, unsigned short* __restrict__ attn_bf)
{
    __shared__ unsigned short Qs[TT * HDIM];          // 4 KB
    __shared__ unsigned short Kc[ATT_CK * HDIM];      // 16 KB  [key][d]
    __shared__ unsigned short Vt[HDIM * ATT_CK];      // 16 KB  [d][key]
    __shared__ float          Sc[TT * ATT_CK];        // 4 KB
    __shared__ unsigned short Pm[TT * ATT_CK];        // 2 KB
    __shared__ float          red_s[256];
    __shared__ float          m_s[TT], l_s[TT], corr_s[TT];

    const int tid = threadIdx.x, lane = tid & 31, wave = tid >> 5;
    const int l16 = lane & 15;
    const int khalf = (lane >= 16) ? 8 : 0;
    const int bh = blockIdx.x;                // b*16 + h
    const int b = bh >> 4, h = bh & 15;
    const long kvbase = (long)bh * STOT * HDIM;
    const float scale = 0.08838834764831845f; // 1/sqrt(128)

    for (int i = tid; i < TT * HDIM; i += 256) {
        int q = i >> 7, d = i & 127;
        Qs[i] = bf16_h(q_ws[(long)(b * TT + q) * DMODEL + h * HDIM + d]);
    }
    if (tid < TT) { m_s[tid] = -1e30f; l_s[tid] = 0.f; corr_s[tid] = 0.f; }
    v8f oacc = {0.f, 0.f, 0.f, 0.f, 0.f, 0.f, 0.f, 0.f};
    __syncthreads();

    const int nchunks = (STOT + ATT_CK - 1) / ATT_CK;   // 65
    for (int c = 0; c < nchunks; ++c) {
        int kstart = c * ATT_CK;
        int kvalid = STOT - kstart; if (kvalid > ATT_CK) kvalid = ATT_CK;

        // prefetch next K/V chunk toward L2 while we work on this one
        if (kstart + ATT_CK < STOT) {
            int pkey = tid >> 2;               // 64 keys, 4 threads/row
            long po = kvbase + (long)(kstart + ATT_CK + pkey) * HDIM + (tid & 3) * 32;
            __builtin_prefetch(Kout + po, 0, 1);
            __builtin_prefetch(Vout + po, 0, 1);
        }

        // load K chunk row-major bf16, V chunk transposed bf16
        for (int i = tid; i < ATT_CK * 32; i += 256) {
            int key = i >> 5, d4 = i & 31;
            if (key < kvalid) {
                float4 kf = *(const float4*)(Kout + kvbase + (long)(kstart + key) * HDIM + d4 * 4);
                Kc[key * HDIM + d4 * 4 + 0] = bf16_h(kf.x);
                Kc[key * HDIM + d4 * 4 + 1] = bf16_h(kf.y);
                Kc[key * HDIM + d4 * 4 + 2] = bf16_h(kf.z);
                Kc[key * HDIM + d4 * 4 + 3] = bf16_h(kf.w);
                float4 vf = *(const float4*)(Vout + kvbase + (long)(kstart + key) * HDIM + d4 * 4);
                Vt[(d4 * 4 + 0) * ATT_CK + key] = bf16_h(vf.x);
                Vt[(d4 * 4 + 1) * ATT_CK + key] = bf16_h(vf.y);
                Vt[(d4 * 4 + 2) * ATT_CK + key] = bf16_h(vf.z);
                Vt[(d4 * 4 + 3) * ATT_CK + key] = bf16_h(vf.w);
            } else {
                Kc[key * HDIM + d4 * 4 + 0] = 0; Kc[key * HDIM + d4 * 4 + 1] = 0;
                Kc[key * HDIM + d4 * 4 + 2] = 0; Kc[key * HDIM + d4 * 4 + 3] = 0;
                Vt[(d4 * 4 + 0) * ATT_CK + key] = 0; Vt[(d4 * 4 + 1) * ATT_CK + key] = 0;
                Vt[(d4 * 4 + 2) * ATT_CK + key] = 0; Vt[(d4 * 4 + 3) * ATT_CK + key] = 0;
            }
        }
        __syncthreads();

        // scores: waves 0..3 each own a 16-key subtile (16q x 16k), K-dim = Dh
        if (wave < 4) {
            v8f sc = {0.f, 0.f, 0.f, 0.f, 0.f, 0.f, 0.f, 0.f};
            int keyl = wave * 16 + l16;
            #pragma unroll
            for (int dc = 0; dc < 4; ++dc) {
                FragBF fa, fb;
                #pragma unroll
                for (int v = 0; v < 8; ++v) {
                    int kk = ((v >= 4) ? 16 : 0) + khalf + (v & 3) * 2;
                    fa.u[v] = *(const unsigned*)(&Qs[l16 * HDIM + dc * 32 + kk]);
                    fb.u[v] = *(const unsigned*)(&Kc[keyl * HDIM + dc * 32 + kk]);
                }
                sc = WMMA_BF16(fa.v, fb.v, sc);
            }
            #pragma unroll
            for (int v = 0; v < 8; ++v) {
                int q = v + ((lane >= 16) ? 8 : 0);
                int kg = kstart + keyl;
                bool ok = (keyl < kvalid) && (kg <= SCACHE + q);   // causal
                Sc[q * ATT_CK + keyl] = ok ? sc[v] * scale : -1e30f;
            }
        }
        __syncthreads();

        // online softmax: 16 threads per query
        int q = tid >> 4, j = tid & 15;
        float tmax = -1e30f;
        #pragma unroll
        for (int jj = 0; jj < ATT_CK / 16; ++jj)
            tmax = fmaxf(tmax, Sc[q * ATT_CK + j * (ATT_CK / 16) + jj]);
        red_s[tid] = tmax; __syncthreads();
        for (int off = 8; off >= 1; off >>= 1) {
            if (j < off) red_s[tid] = fmaxf(red_s[tid], red_s[tid + off]);
            __syncthreads();
        }
        float mold = m_s[q];
        float mnew = fmaxf(mold, red_s[q * 16]);
        float lsum = 0.f;
        #pragma unroll
        for (int jj = 0; jj < ATT_CK / 16; ++jj) {
            int key = j * (ATT_CK / 16) + jj;
            float p = __expf(Sc[q * ATT_CK + key] - mnew);
            Pm[q * ATT_CK + key] = bf16_h(p);
            lsum += p;
        }
        __syncthreads();
        red_s[tid] = lsum; __syncthreads();
        for (int off = 8; off >= 1; off >>= 1) {
            if (j < off) red_s[tid] += red_s[tid + off];
            __syncthreads();
        }
        if (j == 0) {
            float corr = __expf(mold - mnew);
            l_s[q] = l_s[q] * corr + red_s[q * 16];
            m_s[q] = mnew;
            corr_s[q] = corr;
        }
        __syncthreads();

        // O += P x V : each wave owns d-tile [16q x 16d], K-dim = keys
        #pragma unroll
        for (int v = 0; v < 8; ++v)
            oacc[v] *= corr_s[v + ((lane >= 16) ? 8 : 0)];
        int dcol = wave * 16 + l16;
        #pragma unroll
        for (int kc = 0; kc < ATT_CK / 32; ++kc) {
            FragBF fa, fb;
            #pragma unroll
            for (int v = 0; v < 8; ++v) {
                int kk = ((v >= 4) ? 16 : 0) + khalf + (v & 3) * 2;
                fa.u[v] = *(const unsigned*)(&Pm[l16 * ATT_CK + kc * 32 + kk]);
                fb.u[v] = *(const unsigned*)(&Vt[dcol * ATT_CK + kc * 32 + kk]);
            }
            oacc = WMMA_BF16(fa.v, fb.v, oacc);
        }
        __syncthreads();
    }

    #pragma unroll
    for (int v = 0; v < 8; ++v) {
        int q = v + ((lane >= 16) ? 8 : 0);
        int d = wave * 16 + l16;
        float val = oacc[v] / l_s[q];
        attn_bf[(long)(b * TT + q) * DMODEL + h * HDIM + d] = bf16_h(val);
    }
}

// ---------- host-side orchestration ----------
extern "C" void kernel_launch(void* const* d_in, const int* in_sizes, int n_in,
                              void* d_out, int out_size, void* d_ws, size_t ws_size,
                              hipStream_t stream) {
    (void)in_sizes; (void)n_in; (void)out_size; (void)ws_size;
    const float* x       = (const float*)d_in[0];
    const float* k_cache = (const float*)d_in[1];
    const float* v_cache = (const float*)d_in[2];
    const float* ln1_g   = (const float*)d_in[3];
    const float* ln1_b   = (const float*)d_in[4];
    const float* ln2_g   = (const float*)d_in[5];
    const float* ln2_b   = (const float*)d_in[6];
    const float* Wq      = (const float*)d_in[7];
    const float* Wk      = (const float*)d_in[8];
    const float* Wv      = (const float*)d_in[9];
    const float* Wo      = (const float*)d_in[10];
    const float* W1      = (const float*)d_in[11];
    const float* b1      = (const float*)d_in[12];
    const float* W2      = (const float*)d_in[13];
    const float* b2      = (const float*)d_in[14];

    float* xOut = (float*)d_out;                          // [128, 2048]
    float* Kout = xOut + (long)ROWS * DMODEL;             // [8,16,4112,128]
    float* Vout = Kout + (long)BB * NHEADS * STOT * HDIM; // [8,16,4112,128]

    char* ws = (char*)d_ws;
    unsigned short* h_bf    = (unsigned short*)(ws + 0);        // 512 KB
    float*          q_ws    = (float*)(ws + (512 << 10));       // 1 MB
    float*          knew    = (float*)(ws + (1536 << 10));      // 1 MB
    float*          vnew    = (float*)(ws + (2560 << 10));      // 1 MB
    unsigned short* attn_bf = (unsigned short*)(ws + (3584 << 10)); // 512 KB
    float*          x1_ws   = (float*)(ws + (4096 << 10));      // 1 MB
    unsigned short* h2_bf   = (unsigned short*)(ws + (5120 << 10)); // 512 KB
    unsigned short* ffn_bf  = (unsigned short*)(ws + (5632 << 10)); // 2 MB

    // 1) K/V cache -> output concat (the big bandwidth term)
    copy_cache_kernel<<<65536, 256, 0, stream>>>(
        (const float4*)k_cache, (const float4*)v_cache, (float4*)Kout, (float4*)Vout);

    // 2) LN1 -> bf16 activations
    ln_kernel<<<ROWS, 256, 0, stream>>>(x, ln1_g, ln1_b, h_bf);

    // 3) Q/K/V projections (WMMA)
    gemm128_kernel<<<DMODEL / 16, 256, 0, stream>>>(h_bf, Wq, nullptr, nullptr, q_ws, nullptr, DMODEL, DMODEL, 0);
    gemm128_kernel<<<DMODEL / 16, 256, 0, stream>>>(h_bf, Wk, nullptr, nullptr, knew, nullptr, DMODEL, DMODEL, 0);
    gemm128_kernel<<<DMODEL / 16, 256, 0, stream>>>(h_bf, Wv, nullptr, nullptr, vnew, nullptr, DMODEL, DMODEL, 0);

    // 4) place K_new / V_new at s = 4096..4111
    scatter_new_kernel<<<(ROWS * DMODEL) / 256, 256, 0, stream>>>(knew, vnew, Kout, Vout);

    // 5) flash attention over S=4112 (WMMA scores + WMMA P.V)
    attn_kernel<<<BB * NHEADS, 256, 0, stream>>>(q_ws, Kout, Vout, attn_bf);

    // 6) Wo projection + residual x
    gemm128_kernel<<<DMODEL / 16, 256, 0, stream>>>(attn_bf, Wo, nullptr, x, x1_ws, nullptr, DMODEL, DMODEL, 0);

    // 7) LN2 -> bf16
    ln_kernel<<<ROWS, 256, 0, stream>>>(x1_ws, ln2_g, ln2_b, h2_bf);

    // 8) FFN1 + bias + exact GELU -> bf16
    gemm128_kernel<<<DFF / 16, 256, 0, stream>>>(h2_bf, W1, b1, nullptr, nullptr, ffn_bf, DMODEL, DFF, 1);

    // 9) FFN2 + bias + residual -> final x output
    gemm128_kernel<<<DMODEL / 16, 256, 0, stream>>>(ffn_bf, W2, b2, x1_ws, xOut, nullptr, DFF, DMODEL, 0);
}
